// BidirectionalMAMBA_27281632264334
// MI455X (gfx1250) — matrix operand order repaired and checked by reading, not compiled, not run on hardware
//
#include <hip/hip_runtime.h>
#include <hip/hip_bf16.h>
#include <math.h>

// ---------------------------------------------------------------------------
// Problem constants (from reference)
// ---------------------------------------------------------------------------
#define B_SZ  2
#define N_SEQ 1024
#define DM    512          // D_MODEL
#define DI    1024         // D_INNER
#define DST   16           // D_STATE
#define DTR   32           // DT_RANK
#define NL    4            // layers
#define KC    4            // conv kernel
#define TOK   (B_SZ * N_SEQ)   // 2048 tokens
#define KMAX  1024
#define KPAD_OF(K) ((K) + 16)          // half elements; keeps 32B alignment + bank skew
#define KPADMAX (KMAX + 16)

typedef _Float16 half_t;
typedef __attribute__((ext_vector_type(16))) _Float16 v16h;
typedef __attribute__((ext_vector_type(8)))  _Float16 v8h;
typedef __attribute__((ext_vector_type(8)))  float    v8f;

union V16H { v16h v; v8h p[2]; half_t h[16]; };
union V8F  { v8f  v; float  f[8];  };

// ---------------------------------------------------------------------------
// Helpers
// ---------------------------------------------------------------------------
__device__ __forceinline__ float wave_sum(float v) {
#pragma unroll
  for (int off = 16; off > 0; off >>= 1) v += __shfl_xor(v, off, 32);
  return v;
}

__device__ __forceinline__ float silu_f(float x) {
  return x / (1.f + __expf(-x));
}

// ---------------------------------------------------------------------------
// Conversions / copies
// ---------------------------------------------------------------------------
__global__ __launch_bounds__(256) void cvt_f16_kernel(const float* __restrict__ src,
                                                      half_t* __restrict__ dst,
                                                      size_t n) {
  size_t i = (size_t)blockIdx.x * blockDim.x + threadIdx.x;
  if (i < n) dst[i] = (half_t)src[i];
}

__global__ __launch_bounds__(256) void copy_kernel(const float* __restrict__ src,
                                                   float* __restrict__ dst, size_t n) {
  size_t i = (size_t)blockIdx.x * blockDim.x + threadIdx.x;
  if (i < n) dst[i] = src[i];
}

// reverse sequence dimension: o[b, t, c] = x[b, N-1-t, c]
__global__ __launch_bounds__(256) void flip_copy_kernel(const float* __restrict__ src,
                                                        float* __restrict__ dst) {
  size_t i = (size_t)blockIdx.x * blockDim.x + threadIdx.x;
  if (i >= (size_t)TOK * DM) return;
  int c   = (int)(i % DM);
  int tok = (int)(i / DM);
  int t   = tok % N_SEQ;
  int bb  = tok / N_SEQ;
  dst[i] = src[((size_t)bb * N_SEQ + (N_SEQ - 1 - t)) * DM + c];
}

// ---------------------------------------------------------------------------
// LayerNorm (f32 in, f16 out), one block per token row
// ---------------------------------------------------------------------------
__global__ __launch_bounds__(256) void ln_f16_kernel(const float* __restrict__ X,
                                                     const float* __restrict__ g,
                                                     const float* __restrict__ bta,
                                                     half_t* __restrict__ O, int W) {
  __shared__ float sred[8];
  __shared__ float sstat[2];
  const int row = blockIdx.x;
  const float* xr = X + (size_t)row * W;
  const int lane = threadIdx.x & 31, wv = threadIdx.x >> 5;

  float s = 0.f;
  for (int i = threadIdx.x; i < W; i += 256) s += xr[i];
  s = wave_sum(s);
  if (lane == 0) sred[wv] = s;
  __syncthreads();
  if (threadIdx.x == 0) {
    float t = 0.f;
    for (int w2 = 0; w2 < 8; ++w2) t += sred[w2];
    sstat[0] = t / (float)W;
  }
  __syncthreads();
  float m = sstat[0], vs = 0.f;
  for (int i = threadIdx.x; i < W; i += 256) { float d = xr[i] - m; vs += d * d; }
  vs = wave_sum(vs);
  if (lane == 0) sred[wv] = vs;
  __syncthreads();
  if (threadIdx.x == 0) {
    float t = 0.f;
    for (int w2 = 0; w2 < 8; ++w2) t += sred[w2];
    sstat[1] = rsqrtf(t / (float)W + 1e-5f);
  }
  __syncthreads();
  float rs = sstat[1];
  half_t* orow = O + (size_t)row * W;
  for (int i = threadIdx.x; i < W; i += 256)
    orow[i] = (half_t)((xr[i] - m) * rs * g[i] + bta[i]);
}

// Final combine: row = [xf[b,t,:], xb[b,N-1-t,:]] -> LN(1024) -> f16
__global__ __launch_bounds__(256) void concat_ln_f16_kernel(const float* __restrict__ xf,
                                                            const float* __restrict__ xb,
                                                            const float* __restrict__ g,
                                                            const float* __restrict__ bta,
                                                            half_t* __restrict__ O) {
  __shared__ float sx[2 * DM];
  __shared__ float sred[8];
  __shared__ float sstat[2];
  const int row = blockIdx.x;
  const int bb = row / N_SEQ, t = row % N_SEQ;
  const float* p1 = xf + (size_t)row * DM;
  const float* p2 = xb + ((size_t)bb * N_SEQ + (N_SEQ - 1 - t)) * DM;
  for (int i = threadIdx.x; i < DM; i += 256) { sx[i] = p1[i]; sx[DM + i] = p2[i]; }
  __syncthreads();

  const int W = 2 * DM;
  const int lane = threadIdx.x & 31, wv = threadIdx.x >> 5;
  float s = 0.f;
  for (int i = threadIdx.x; i < W; i += 256) s += sx[i];
  s = wave_sum(s);
  if (lane == 0) sred[wv] = s;
  __syncthreads();
  if (threadIdx.x == 0) {
    float tt = 0.f;
    for (int w2 = 0; w2 < 8; ++w2) tt += sred[w2];
    sstat[0] = tt / (float)W;
  }
  __syncthreads();
  float m = sstat[0], vs = 0.f;
  for (int i = threadIdx.x; i < W; i += 256) { float d = sx[i] - m; vs += d * d; }
  vs = wave_sum(vs);
  if (lane == 0) sred[wv] = vs;
  __syncthreads();
  if (threadIdx.x == 0) {
    float tt = 0.f;
    for (int w2 = 0; w2 < 8; ++w2) tt += sred[w2];
    sstat[1] = rsqrtf(tt / (float)W + 1e-5f);
  }
  __syncthreads();
  float rs = sstat[1];
  half_t* orow = O + (size_t)row * W;
  for (int i = threadIdx.x; i < W; i += 256)
    orow[i] = (half_t)((sx[i] - m) * rs * g[i] + bta[i]);
}

// ---------------------------------------------------------------------------
// WMMA GEMM: C[M,N] = epilogue( A[M,K](f16,row) x B[K,N](f16,row) )
//
// Block tile = 128 (M) x 16 (N); 8 waves stacked in M share one B strip.
// B strip (K x 16) is staged once into LDS *transposed* to [n][Kpad] so each
// lane's 16-half fragment is two contiguous ds_load_b128s.  A fragments are
// direct global b128 pairs.  One v_wmma_f32_16x16x32_f16 per 32-K step.
// Grid is sized exactly -> EXEC is all-ones for every WMMA.
//
// epi: 0=none  1=softplus(acc+bias[col])  2=acc+resid  3=gelu(acc+bias[col])
// ---------------------------------------------------------------------------
__global__ __launch_bounds__(256) void gemm_wmma_f16(const half_t* __restrict__ A,
                                                     const half_t* __restrict__ Bw,
                                                     float* __restrict__ C,
                                                     int M, int N, int K,
                                                     const float* __restrict__ resid,
                                                     const float* __restrict__ bias,
                                                     int epi) {
  __shared__ half_t sB[16 * KPADMAX];   // B strip, [n][Kpad] transposed

  const int ntN  = N >> 4;
  const int tn   = blockIdx.x % ntN;          // N tile of the whole block
  const int tmB  = blockIdx.x / ntN;          // block-of-128 row index
  const int lane = threadIdx.x & 31;
  const int wv   = threadIdx.x >> 5;
  const int Kpad = KPAD_OF(K);

  // ---- stage B strip (K x 16) -> LDS transposed [n][Kpad] ----
  {
    const half_t* Bsrc = Bw + tn * 16;
    const int total = K * 16;
    for (int idx = threadIdx.x; idx < total; idx += 256) {
      int r = idx >> 4;     // k
      int c = idx & 15;     // n within tile
      sB[c * Kpad + r] = Bsrc[(size_t)r * N + c];
    }
  }
  __syncthreads();

  // ---- per-wave 16x16 tile ----
  const int tm  = tmB * 8 + wv;
  const int mr  = lane & 15;            // A row in tile / C column in tile
  const int kgA = (lane >> 4) << 3;     // A K sub-offset: 0 or 8
  const int kgB = (lane >> 4) << 3;     // B K sub-offset: 0 or 8

  V8F acc = {};
  const half_t* Ap0 = A + (size_t)(tm * 16 + mr) * K + kgA;
  const half_t* Bl0 = &sB[(lane & 15) * Kpad + kgB];

  for (int k0 = 0; k0 < K; k0 += 32) {
    V16H a;
    a.p[0] = *(const v8h*)(Ap0 + k0);        // K = kgA .. kgA+7
    a.p[1] = *(const v8h*)(Ap0 + k0 + 16);   // K = 16+kgA .. 16+kgA+7
    __builtin_prefetch(Ap0 + k0 + 64, 0, 0); // next A block -> global_prefetch_b8

    V16H b;                                   // K = kgB..kgB+7, 16+kgB..16+kgB+7 from LDS
    b.p[0] = *(const v8h*)(Bl0 + k0);
    b.p[1] = *(const v8h*)(Bl0 + k0 + 16);

    acc.v = __builtin_amdgcn_wmma_f32_16x16x32_f16(
        false, a.v, false, b.v, (short)0, acc.v, false, false);
  }

  const int col   = tn * 16 + (lane & 15);
  const int rbase = tm * 16 + ((lane >> 4) << 3);
#pragma unroll
  for (int r = 0; r < 8; ++r) {
    int row = rbase + r;
    float v = acc.f[r];
    if (epi == 1) {
      v += bias[col];
      v = (v > 20.f) ? v : log1pf(__expf(v));  // softplus
    } else if (epi == 2) {
      v += resid[(size_t)row * N + col];
    } else if (epi == 3) {
      v += bias[col];
      v = 0.5f * v * (1.f + erff(v * 0.70710678118654752f));  // exact GELU
    }
    C[(size_t)row * N + col] = v;
  }
}

// ---------------------------------------------------------------------------
// Depthwise causal conv (K=4) + bias + SiLU; reads u half of uz buffer.
// ---------------------------------------------------------------------------
__global__ __launch_bounds__(256) void conv_silu_kernel(const float* __restrict__ uz,
                                                        const float* __restrict__ cw,
                                                        const float* __restrict__ cb,
                                                        float* __restrict__ uf,
                                                        half_t* __restrict__ uh) {
  size_t idx = (size_t)blockIdx.x * blockDim.x + threadIdx.x;
  if (idx >= (size_t)TOK * DI) return;
  int d   = (int)(idx % DI);
  int tok = (int)(idx / DI);
  int t   = tok % N_SEQ;
  int bb  = tok / N_SEQ;
  float acc = cb[d];
#pragma unroll
  for (int k = 0; k < KC; ++k) {
    int ts = t - (KC - 1) + k;
    if (ts >= 0)
      acc += cw[d * KC + k] * uz[((size_t)bb * N_SEQ + ts) * (2 * DI) + d];
  }
  float sv = silu_f(acc);
  uf[idx] = sv;
  uh[idx] = (half_t)sv;
}

// Extract first DTR columns of xdbl (row stride 64) as f16
__global__ __launch_bounds__(256) void xdbl_f16_kernel(const float* __restrict__ xdbl,
                                                       half_t* __restrict__ o) {
  size_t idx = (size_t)blockIdx.x * blockDim.x + threadIdx.x;
  if (idx >= (size_t)TOK * DTR) return;
  int c = (int)(idx % DTR);
  size_t r = idx / DTR;
  o[idx] = (half_t)xdbl[r * (DTR + 2 * DST) + c];
}

// ---------------------------------------------------------------------------
// Selective scan: one thread per (b, d); states + A in registers.
// ---------------------------------------------------------------------------
__global__ __launch_bounds__(256) void scan_kernel(const float* __restrict__ dt,
                                                   const float* __restrict__ u,
                                                   const float* __restrict__ xdbl,
                                                   const float* __restrict__ Alog,
                                                   float* __restrict__ ys) {
  int idx = blockIdx.x * blockDim.x + threadIdx.x;   // b*DI + d
  if (idx >= B_SZ * DI) return;
  int d  = idx % DI;
  int bb = idx / DI;

  float An[DST], h[DST];
#pragma unroll
  for (int s = 0; s < DST; ++s) {
    An[s] = -__expf(Alog[(size_t)d * DST + s]);      // A = -exp(Alog)
    h[s]  = 0.f;
  }

  const size_t base = (size_t)bb * N_SEQ;
  for (int t = 0; t < N_SEQ; ++t) {
    size_t tok = base + t;
    float dtv = dt[tok * DI + d];
    float du  = dtv * u[tok * DI + d];
    const float* xr = xdbl + tok * (DTR + 2 * DST);
    float y = 0.f;
#pragma unroll
    for (int s = 0; s < DST; ++s) {
      float dA = __expf(dtv * An[s]);
      h[s] = dA * h[s] + du * xr[DTR + s];           // Bm
      y += h[s] * xr[DTR + DST + s];                 // Cm
    }
    ys[tok * DI + d] = y;
  }
}

// y = (yscan + u*D) * silu(z), f16 out
__global__ __launch_bounds__(256) void gate_kernel(const float* __restrict__ ys,
                                                   const float* __restrict__ uf,
                                                   const float* __restrict__ uz,
                                                   const float* __restrict__ Dv,
                                                   half_t* __restrict__ y16) {
  size_t idx = (size_t)blockIdx.x * blockDim.x + threadIdx.x;
  if (idx >= (size_t)TOK * DI) return;
  int d = (int)(idx % DI);
  size_t tok = idx / DI;
  float y = ys[idx] + uf[idx] * Dv[d];
  float z = uz[tok * (2 * DI) + DI + d];
  y16[idx] = (half_t)(y * silu_f(z));
}

// ---------------------------------------------------------------------------
// Host orchestration
// ---------------------------------------------------------------------------
extern "C" void kernel_launch(void* const* d_in, const int* in_sizes, int n_in,
                              void* d_out, int out_size, void* d_ws, size_t ws_size,
                              hipStream_t stream) {
  (void)in_sizes; (void)n_in; (void)out_size; (void)ws_size;

  const float* x = (const float*)d_in[0];
  // per-direction params: 0 ln_g, 1 ln_b, 2 Win, 3 conv_w, 4 conv_b,
  //                       5 Wx, 6 Wdt, 7 bdt, 8 Alog, 9 D, 10 Wout
  const float* p[2][11];
  for (int dir = 0; dir < 2; ++dir)
    for (int j = 0; j < 11; ++j)
      p[dir][j] = (const float*)d_in[1 + dir * 11 + j];
  const float* cmb_ln_g = (const float*)d_in[23];
  const float* cmb_ln_b = (const float*)d_in[24];
  const float* cmb_W    = (const float*)d_in[25];
  const float* cmb_b    = (const float*)d_in[26];

  char* base = (char*)d_ws;
  size_t off = 0;
  auto alloc = [&](size_t bytes) -> void* {
    void* r = base + off;
    off = (off + bytes + 255) & ~(size_t)255;
    return r;
  };
  const int DXB = DTR + 2 * DST;  // 64
  half_t* Win16  = (half_t*)alloc((size_t)2 * NL * DM * 2 * DI * sizeof(half_t));
  half_t* Wx16   = (half_t*)alloc((size_t)2 * NL * DI * DXB * sizeof(half_t));
  half_t* Wdt16  = (half_t*)alloc((size_t)2 * NL * DTR * DI * sizeof(half_t));
  half_t* Wout16 = (half_t*)alloc((size_t)2 * NL * DI * DM * sizeof(half_t));
  half_t* cmbW16 = (half_t*)alloc((size_t)DI * DM * sizeof(half_t));
  float* xbuf0   = (float*)alloc((size_t)TOK * DM * sizeof(float));
  float* xbuf1   = (float*)alloc((size_t)TOK * DM * sizeof(float));
  half_t* h16    = (half_t*)alloc((size_t)TOK * DM * sizeof(half_t));
  float*  uz     = (float*)alloc((size_t)TOK * 2 * DI * sizeof(float));
  float*  uf     = (float*)alloc((size_t)TOK * DI * sizeof(float));
  half_t* u16    = (half_t*)alloc((size_t)TOK * DI * sizeof(half_t));
  float*  xdbl   = (float*)alloc((size_t)TOK * DXB * sizeof(float));
  half_t* xdbl16 = (half_t*)alloc((size_t)TOK * DTR * sizeof(half_t));
  float*  dtb    = (float*)alloc((size_t)TOK * DI * sizeof(float));
  float*  ys     = (float*)alloc((size_t)TOK * DI * sizeof(float));
  half_t* y16    = (half_t*)alloc((size_t)TOK * DI * sizeof(half_t));
  half_t* hc16   = (half_t*)alloc((size_t)TOK * DI * sizeof(half_t));
  float* xbuf[2] = { xbuf0, xbuf1 };

  auto cvt = [&](const float* src, half_t* dst, size_t n) {
    cvt_f16_kernel<<<(unsigned)((n + 255) / 256), 256, 0, stream>>>(src, dst, n);
  };
  for (int dir = 0; dir < 2; ++dir) {
    cvt(p[dir][2],  Win16  + (size_t)dir * NL * DM * 2 * DI, (size_t)NL * DM * 2 * DI);
    cvt(p[dir][5],  Wx16   + (size_t)dir * NL * DI * DXB,    (size_t)NL * DI * DXB);
    cvt(p[dir][6],  Wdt16  + (size_t)dir * NL * DTR * DI,    (size_t)NL * DTR * DI);
    cvt(p[dir][10], Wout16 + (size_t)dir * NL * DI * DM,     (size_t)NL * DI * DM);
  }
  cvt(cmb_W, cmbW16, (size_t)DI * DM);

  auto gemm = [&](const half_t* A, const half_t* Bw, float* C, int M, int N, int K,
                  const float* resid, const float* bias, int epi) {
    int blocks = (M / 128) * (N / 16);   // M is always a multiple of 128 here
    gemm_wmma_f16<<<blocks, 256, 0, stream>>>(A, Bw, C, M, N, K, resid, bias, epi);
  };

  copy_kernel<<<(TOK * DM) / 256, 256, 0, stream>>>(x, xbuf[0], (size_t)TOK * DM);
  flip_copy_kernel<<<(TOK * DM) / 256, 256, 0, stream>>>(x, xbuf[1]);

  for (int dir = 0; dir < 2; ++dir) {
    float* xc = xbuf[dir];
    for (int l = 0; l < NL; ++l) {
      const half_t* Wi  = Win16  + ((size_t)dir * NL + l) * DM * 2 * DI;
      const half_t* Wxp = Wx16   + ((size_t)dir * NL + l) * DI * DXB;
      const half_t* Wdp = Wdt16  + ((size_t)dir * NL + l) * DTR * DI;
      const half_t* Wop = Wout16 + ((size_t)dir * NL + l) * DI * DM;

      ln_f16_kernel<<<TOK, 256, 0, stream>>>(xc, p[dir][0] + l * DM,
                                             p[dir][1] + l * DM, h16, DM);
      gemm(h16, Wi, uz, TOK, 2 * DI, DM, nullptr, nullptr, 0);
      conv_silu_kernel<<<(TOK * DI) / 256, 256, 0, stream>>>(
          uz, p[dir][3] + (size_t)l * DI * KC, p[dir][4] + l * DI, uf, u16);
      gemm(u16, Wxp, xdbl, TOK, DXB, DI, nullptr, nullptr, 0);
      xdbl_f16_kernel<<<(TOK * DTR) / 256, 256, 0, stream>>>(xdbl, xdbl16);
      gemm(xdbl16, Wdp, dtb, TOK, DI, DTR, nullptr, p[dir][7] + l * DI, 1);
      scan_kernel<<<(B_SZ * DI) / 256, 256, 0, stream>>>(
          dtb, uf, xdbl, p[dir][8] + (size_t)l * DI * DST, ys);
      gate_kernel<<<(TOK * DI) / 256, 256, 0, stream>>>(ys, uf, uz,
                                                        p[dir][9] + l * DI, y16);
      gemm(y16, Wop, xc, TOK, DM, DI, xc, nullptr, 2);
    }
  }

  concat_ln_f16_kernel<<<TOK, 256, 0, stream>>>(xbuf[0], xbuf[1],
                                                cmb_ln_g, cmb_ln_b, hc16);
  gemm(hc16, cmbW16, (float*)d_out, TOK, DM, DI, nullptr, cmb_b, 3);
}
